// VisualPromptEncoder_45672682226224
// MI455X (gfx1250) — matrix operand order repaired and checked by reading, not compiled, not run on hardware
//
#include <hip/hip_runtime.h>
#include <hip/hip_bf16.h>

// ---------------- types ----------------
typedef __attribute__((ext_vector_type(16))) __bf16 v16bf;
typedef __attribute__((ext_vector_type(8)))  float  v8f;

#define BATCH   8
#define KPTS    100
#define QTOT    101          // 100 points + 1 global token
#define GROWS   (BATCH*QTOT) // 808
#define GPAD    816          // padded to multiple of 16
#define CDIM    256
#define VTOT    13294        // 100*100 + 50*50 + 25*25 + 13*13
#define MV      (BATCH*VTOT) // 106352 = 16*6647

// =====================================================================
// Weight pre-pack: W[k][n] (row-major KxN, f32) -> bf16 fragments laid out
// so a wave's B-operand for tile (tn, kb) is contiguous per lane:
//   Wp[((tn*KB + kb)*32 + lane)*16 + i] = W[kb*32 + 16*(lane>>4) + i][tn*16 + (lane&15)]
// =====================================================================
__global__ void pack_w_kernel(const float* __restrict__ W, __bf16* __restrict__ Wp,
                              int K, int N) {
    int t = blockIdx.x * blockDim.x + threadIdx.x;
    if (t >= K * N) return;
    int i    = t & 15;
    int lane = (t >> 4) & 31;
    int rest = t >> 9;
    int KB   = K >> 5;
    int kb   = rest % KB;
    int tn   = rest / KB;
    int k = kb * 32 + 16 * (lane >> 4) + i;
    int n = tn * 16 + (lane & 15);
    Wp[t] = (__bf16)W[(size_t)k * N + n];
}

// f32 activation [Mvalid,K] -> bf16 [Mpad,K], zero-filled pad rows
__global__ void pack_act_kernel(const float* __restrict__ A, __bf16* __restrict__ Ap,
                                int Mvalid, int Mpad, int K) {
    int t = blockIdx.x * blockDim.x + threadIdx.x;
    if (t >= Mpad * K) return;
    int row = t / K;
    Ap[t] = (row < Mvalid) ? (__bf16)A[t] : (__bf16)0.0f;
}

// =====================================================================
// WMMA GEMM: out[M,N] f32 = A[M,K] bf16 (row major) x Wp (packed) + bias
// One wave per 16x16 output tile; K consumed 32 at a time with
// v_wmma_f32_16x16x32_bf16. M,N multiples of 16; K multiple of 32.
// =====================================================================
__global__ void gemm_bf16_kernel(const __bf16* __restrict__ A,
                                 const __bf16* __restrict__ Wp,
                                 const float* __restrict__ bias,
                                 float* __restrict__ out,
                                 int M, int N, int K, int relu) {
    const int lane = threadIdx.x & 31;
    const int wave = threadIdx.x >> 5;
    const int ntn  = N >> 4;
    const int tile = blockIdx.x * (blockDim.x >> 5) + wave;
    if (tile >= (M >> 4) * ntn) return;        // wave-uniform: EXEC stays all-ones
    const int tm   = tile / ntn;
    const int tn   = tile - tm * ntn;
    const int half = lane >> 4;
    const int l16  = lane & 15;
    const int row  = (tm << 4) + l16;
    const int KB   = K >> 5;

    v8f acc = {};
    const __bf16* ap = A  + (size_t)row * K + 16 * half;
    const __bf16* bp = Wp + (((size_t)tn * KB) * 32 + lane) * 16;

    for (int kb = 0; kb < KB; ++kb) {
        v16bf a, b;
        ((uint4*)&a)[0] = *(const uint4*)(ap);
        ((uint4*)&a)[1] = *(const uint4*)(ap + 8);
        ((uint4*)&b)[0] = ((const uint4*)bp)[0];
        ((uint4*)&b)[1] = ((const uint4*)bp)[1];
        acc = __builtin_amdgcn_wmma_f32_16x16x32_bf16(
                  false, a, false, b, (short)0, acc, false, false);
        ap += 32;
        bp += 32 * 16;
    }

    const int n = (tn << 4) + l16;
    const float bv = bias ? bias[n] : 0.0f;
    #pragma unroll
    for (int r = 0; r < 8; ++r) {
        float v = acc[r] + bv;
        if (relu) v = fmaxf(v, 0.0f);
        out[(size_t)((tm << 4) + r + 8 * half) * N + n] = v;
    }
}

// =====================================================================
// Build A for vproj: flatten [B,C,h,w] -> [B, v, C], add pos, cast bf16
// =====================================================================
__global__ void build_av_kernel(const float* __restrict__ f0, const float* __restrict__ p0,
                                const float* __restrict__ f1, const float* __restrict__ p1,
                                const float* __restrict__ f2, const float* __restrict__ p2,
                                const float* __restrict__ f3, const float* __restrict__ p3,
                                __bf16* __restrict__ Av) {
    int t = blockIdx.x * blockDim.x + threadIdx.x;
    if (t >= MV * CDIM) return;
    int c   = t & 255;
    int row = t >> 8;
    int b   = row / VTOT;
    int v   = row - b * VTOT;
    const float *f, *p; int W, base;
    if (v < 10000)      { f = f0; p = p0; W = 100; base = 0; }
    else if (v < 12500) { f = f1; p = p1; W = 50;  base = 10000; }
    else if (v < 13125) { f = f2; p = p2; W = 25;  base = 12500; }
    else                { f = f3; p = p3; W = 13;  base = 13125; }
    int local = v - base;
    int y = local / W, x = local - y * W;
    size_t off = (((size_t)b * CDIM + c) * W + y) * W + x;  // H==W all levels
    Av[t] = (__bf16)(f[off] + p[off]);
}

// pos MLP layer 1: x1 = relu(points @ pw0 + pb0), rows = B*KPTS = 800
__global__ void pos_l1_kernel(const float* __restrict__ pts, const float* __restrict__ pw0,
                              const float* __restrict__ pb0, float* __restrict__ x1) {
    int t = blockIdx.x * blockDim.x + threadIdx.x;
    if (t >= BATCH * KPTS * CDIM) return;
    int c = t & 255, row = t >> 8;
    float v = pts[row * 2] * pw0[c] + pts[row * 2 + 1] * pw0[CDIM + c] + pb0[c];
    x1[t] = fmaxf(v, 0.0f);
}

// q_in = query + query_pos (content + pos_embed / global token)
__global__ void build_qin_kernel(const float* __restrict__ pos_embed,
                                 const float* __restrict__ pc, const float* __restrict__ gc,
                                 const float* __restrict__ gp, float* __restrict__ q_in) {
    int t = blockIdx.x * blockDim.x + threadIdx.x;
    if (t >= GROWS * CDIM) return;
    int c = t & 255, g = t >> 8;
    int b = g / QTOT, q = g - b * QTOT;
    q_in[t] = (q < KPTS) ? (pc[c] + pos_embed[(b * KPTS + q) * CDIM + c])
                         : (gc[c] + gp[c]);
}

// softmax over the 16 (L*P) attention weights per (g, head); in place
__global__ void softmax_attw_kernel(float* __restrict__ aw) {
    int t = blockIdx.x * blockDim.x + threadIdx.x;
    if (t >= GROWS * 8) return;
    int g = t >> 3, h = t & 7;
    float* p = aw + (size_t)g * 128 + h * 16;
    float v[16], mx = -1e30f;
    #pragma unroll
    for (int i = 0; i < 16; ++i) { v[i] = p[i]; mx = fmaxf(mx, v[i]); }
    float s = 0.0f;
    #pragma unroll
    for (int i = 0; i < 16; ++i) { v[i] = __expf(v[i] - mx); s += v[i]; }
    float inv = 1.0f / s;
    #pragma unroll
    for (int i = 0; i < 16; ++i) p[i] = v[i] * inv;
}

// =====================================================================
// Deformable sampling: thread per (g, head, d) -> msda[g, h*32+d]
// =====================================================================
__global__ void sample_kernel(const float* __restrict__ val,   // [B, VTOT, 256]
                              const float* __restrict__ offs,  // [g, 256]
                              const float* __restrict__ aw,    // [g, 128] softmaxed
                              const float* __restrict__ pts,   // [B, 100, 2]
                              float* __restrict__ out) {
    int t = blockIdx.x * blockDim.x + threadIdx.x;
    if (t >= GROWS * 256) return;
    int d = t & 31, h = (t >> 5) & 7, g = t >> 8;
    int b = g / QTOT, q = g - b * QTOT;
    float rx, ry;
    if (q < KPTS) { rx = pts[(b * KPTS + q) * 2]; ry = pts[(b * KPTS + q) * 2 + 1]; }
    else          { rx = 0.5f; ry = 0.5f; }
    const int lW[4]   = {100, 50, 25, 13};
    const int lB[4]   = {0, 10000, 12500, 13125};
    float acc = 0.0f;
    #pragma unroll
    for (int l = 0; l < 4; ++l) {
        const float Wl = (float)lW[l];
        const int   Wi = lW[l];
        #pragma unroll
        for (int p = 0; p < 4; ++p) {
            int oi = ((h * 4 + l) * 4 + p) * 2;
            float ox = offs[(size_t)g * 256 + oi];
            float oy = offs[(size_t)g * 256 + oi + 1];
            float wat = aw[(size_t)g * 128 + h * 16 + l * 4 + p];
            float x = (rx + ox / Wl) * Wl - 0.5f;   // H==W per level
            float y = (ry + oy / Wl) * Wl - 0.5f;
            float x0 = floorf(x), y0 = floorf(y);
            float fx = x - x0, fy = y - y0;
            #pragma unroll
            for (int dx = 0; dx < 2; ++dx) {
                #pragma unroll
                for (int dy = 0; dy < 2; ++dy) {
                    float xi = x0 + dx, yi = y0 + dy;
                    bool valid = (xi >= 0.f) && (xi < Wl) && (yi >= 0.f) && (yi < Wl);
                    int xc = (int)fminf(fmaxf(xi, 0.f), Wl - 1.f);
                    int yc = (int)fminf(fmaxf(yi, 0.f), Wl - 1.f);
                    float w = (dx ? fx : 1.f - fx) * (dy ? fy : 1.f - fy);
                    float v = val[(size_t)(b * VTOT + lB[l] + yc * Wi + xc) * 256 + h * 32 + d];
                    acc += valid ? (wat * w * v) : 0.0f;
                }
            }
        }
    }
    out[(size_t)g * 256 + h * 32 + d] = acc;
}

// =====================================================================
// Self-attention: one block per (b, head); K/V cached in LDS
// =====================================================================
__global__ void self_attn_kernel(const float* __restrict__ qkv,  // [g, 768]
                                 float* __restrict__ o_attn) {   // [g, 256]
    const int b = blockIdx.x >> 3, h = blockIdx.x & 7;
    __shared__ float sk[QTOT * 32];
    __shared__ float sv[QTOT * 32];
    for (int idx = threadIdx.x; idx < QTOT * 32; idx += blockDim.x) {
        int r = idx >> 5, d = idx & 31;
        size_t base = (size_t)(b * QTOT + r) * 768 + h * 32 + d;
        sk[idx] = qkv[base + 256];
        sv[idx] = qkv[base + 512];
    }
    __syncthreads();
    int qi = threadIdx.x;
    if (qi >= QTOT) return;
    float qv[32];
    #pragma unroll
    for (int d = 0; d < 32; ++d)
        qv[d] = qkv[(size_t)(b * QTOT + qi) * 768 + h * 32 + d];
    const float scale = 0.17677669529663687f;  // 1/sqrt(32)
    float mx = -1e30f;
    for (int k = 0; k < QTOT; ++k) {
        float s = 0.f;
        #pragma unroll
        for (int d = 0; d < 32; ++d) s += qv[d] * sk[k * 32 + d];
        mx = fmaxf(mx, s * scale);
    }
    float sum = 0.f, o[32];
    #pragma unroll
    for (int d = 0; d < 32; ++d) o[d] = 0.f;
    for (int k = 0; k < QTOT; ++k) {
        float s = 0.f;
        #pragma unroll
        for (int d = 0; d < 32; ++d) s += qv[d] * sk[k * 32 + d];
        float w = __expf(s * scale - mx);
        sum += w;
        #pragma unroll
        for (int d = 0; d < 32; ++d) o[d] += w * sv[k * 32 + d];
    }
    float inv = 1.0f / sum;
    #pragma unroll
    for (int d = 0; d < 32; ++d)
        o_attn[(size_t)(b * QTOT + qi) * 256 + h * 32 + d] = o[d] * inv;
}

// =====================================================================
// LayerNorms (block per row, 256 threads)
// =====================================================================
__device__ __forceinline__ float block_ln(float x, const float* g, const float* bt,
                                          int c, float* red) {
    red[c] = x; __syncthreads();
    for (int s = 128; s > 0; s >>= 1) { if (c < s) red[c] += red[c + s]; __syncthreads(); }
    float mean = red[0] * (1.0f / 256.0f);
    __syncthreads();
    float d = x - mean;
    red[c] = d * d; __syncthreads();
    for (int s = 128; s > 0; s >>= 1) { if (c < s) red[c] += red[c + s]; __syncthreads(); }
    float var = red[0] * (1.0f / 256.0f);
    return d * rsqrtf(var + 1e-5f) * g[c] + bt[c];
}

// q1 = LN(query_content + oproj_out + q_in)
__global__ void ln1_kernel(const float* __restrict__ oproj_out, const float* __restrict__ q_in,
                           const float* __restrict__ pc, const float* __restrict__ gc,
                           const float* __restrict__ g, const float* __restrict__ bt,
                           float* __restrict__ out) {
    __shared__ float red[256];
    int row = blockIdx.x, c = threadIdx.x;
    int q = row % QTOT;
    float content = (q < KPTS) ? pc[c] : gc[c];
    float x = content + oproj_out[(size_t)row * 256 + c] + q_in[(size_t)row * 256 + c];
    out[(size_t)row * 256 + c] = block_ln(x, g, bt, c, red);
}

// out = LN(a + b2)
__global__ void ln_add_kernel(const float* __restrict__ a, const float* __restrict__ b2,
                              const float* __restrict__ g, const float* __restrict__ bt,
                              float* __restrict__ out) {
    __shared__ float red[256];
    int row = blockIdx.x, c = threadIdx.x;
    float x = a[(size_t)row * 256 + c] + b2[(size_t)row * 256 + c];
    out[(size_t)row * 256 + c] = block_ln(x, g, bt, c, red);
}

// final: LN(q2 + ffn2) at the global-token rows only -> d_out[B,256]
__global__ void ln3_out_kernel(const float* __restrict__ q2, const float* __restrict__ ffn2,
                               const float* __restrict__ g, const float* __restrict__ bt,
                               float* __restrict__ out) {
    __shared__ float red[256];
    int b = blockIdx.x, c = threadIdx.x;
    int row = b * QTOT + KPTS;   // last token
    float x = q2[(size_t)row * 256 + c] + ffn2[(size_t)row * 256 + c];
    out[(size_t)b * 256 + c] = block_ln(x, g, bt, c, red);
}

// ---------------- host side ----------------
static inline void launch_pack_w(const float* W, __bf16* Wp, int K, int N, hipStream_t s) {
    int total = K * N;
    pack_w_kernel<<<(total + 255) / 256, 256, 0, s>>>(W, Wp, K, N);
}
static inline void launch_pack_act(const float* A, __bf16* Ap, int Mvalid, int Mpad, int K,
                                   hipStream_t s) {
    int total = Mpad * K;
    pack_act_kernel<<<(total + 255) / 256, 256, 0, s>>>(A, Ap, Mvalid, Mpad, K);
}
static inline void launch_gemm(const __bf16* A, const __bf16* Wp, const float* bias,
                               float* out, int M, int N, int K, int relu, hipStream_t s) {
    int tiles = (M / 16) * (N / 16);
    gemm_bf16_kernel<<<(tiles + 7) / 8, 256, 0, s>>>(A, Wp, bias, out, M, N, K, relu);
}

extern "C" void kernel_launch(void* const* d_in, const int* in_sizes, int n_in,
                              void* d_out, int out_size, void* d_ws, size_t ws_size,
                              hipStream_t stream) {
    // inputs (setup_inputs dict order)
    const float* feat[4] = {(const float*)d_in[0], (const float*)d_in[2],
                            (const float*)d_in[4], (const float*)d_in[6]};
    const float* posf[4] = {(const float*)d_in[1], (const float*)d_in[3],
                            (const float*)d_in[5], (const float*)d_in[7]};
    const float* points  = (const float*)d_in[8];
    const float* pw0 = (const float*)d_in[9],  *pb0 = (const float*)d_in[10];
    const float* pw1 = (const float*)d_in[11], *pb1 = (const float*)d_in[12];
    const float* pw2 = (const float*)d_in[13], *pb2 = (const float*)d_in[14];
    const float* pc  = (const float*)d_in[15];
    const float* gc  = (const float*)d_in[16];
    const float* gp  = (const float*)d_in[17];
    const float* vproj_w = (const float*)d_in[18], *vproj_b = (const float*)d_in[19];
    const float* off_w   = (const float*)d_in[20], *off_b   = (const float*)d_in[21];
    const float* attw_w  = (const float*)d_in[22], *attw_b  = (const float*)d_in[23];
    const float* oproj_w = (const float*)d_in[24], *oproj_b = (const float*)d_in[25];
    const float* ln1_g = (const float*)d_in[26], *ln1_b = (const float*)d_in[27];
    const float* sa_in_w = (const float*)d_in[28], *sa_in_b = (const float*)d_in[29];
    const float* sa_out_w = (const float*)d_in[30], *sa_out_b = (const float*)d_in[31];
    const float* ln2_g = (const float*)d_in[32], *ln2_b = (const float*)d_in[33];
    const float* fw0 = (const float*)d_in[34], *fb0 = (const float*)d_in[35];
    const float* fw1 = (const float*)d_in[36], *fb1 = (const float*)d_in[37];
    const float* ln3_g = (const float*)d_in[38], *ln3_b = (const float*)d_in[39];

    // workspace bump allocator (256B aligned)
    size_t off = 0;
    auto alloc = [&](size_t bytes) -> void* {
        void* p = (void*)((char*)d_ws + off);
        off += (bytes + 255) & ~(size_t)255;
        return p;
    };
    const size_t F = sizeof(float), BF = sizeof(__bf16);

    __bf16* Av      = (__bf16*)alloc((size_t)MV * CDIM * BF);
    float*  val     = (float*) alloc((size_t)MV * CDIM * F);
    __bf16* Wp_vproj= (__bf16*)alloc(256 * 256 * BF);
    __bf16* Wp_pw1  = (__bf16*)alloc(256 * 256 * BF);
    __bf16* Wp_pw2  = (__bf16*)alloc(256 * 256 * BF);
    __bf16* Wp_off  = (__bf16*)alloc(256 * 256 * BF);
    __bf16* Wp_attw = (__bf16*)alloc(256 * 128 * BF);
    __bf16* Wp_oproj= (__bf16*)alloc(256 * 256 * BF);
    __bf16* Wp_sain = (__bf16*)alloc(256 * 768 * BF);
    __bf16* Wp_saout= (__bf16*)alloc(256 * 256 * BF);
    __bf16* Wp_fw0  = (__bf16*)alloc(256 * 1024 * BF);
    __bf16* Wp_fw1  = (__bf16*)alloc(1024 * 256 * BF);
    float*  x1      = (float*) alloc(800 * 256 * F);
    __bf16* x1b     = (__bf16*)alloc(800 * 256 * BF);
    float*  x2      = (float*) alloc(800 * 256 * F);
    __bf16* x2b     = (__bf16*)alloc(800 * 256 * BF);
    float*  pos_emb = (float*) alloc(800 * 256 * F);
    float*  q_in    = (float*) alloc(GPAD * 256 * F);
    __bf16* q_inb   = (__bf16*)alloc(GPAD * 256 * BF);
    float*  offs    = (float*) alloc(GPAD * 256 * F);
    float*  attwv   = (float*) alloc(GPAD * 128 * F);
    float*  msda    = (float*) alloc(GPAD * 256 * F);
    __bf16* msdab   = (__bf16*)alloc(GPAD * 256 * BF);
    float*  oproj_o = (float*) alloc(GPAD * 256 * F);
    float*  q1      = (float*) alloc(GPAD * 256 * F);
    __bf16* q1b     = (__bf16*)alloc(GPAD * 256 * BF);
    float*  qkv     = (float*) alloc(GPAD * 768 * F);
    float*  o_attn  = (float*) alloc(GPAD * 256 * F);
    __bf16* o_attnb = (__bf16*)alloc(GPAD * 256 * BF);
    float*  sa_o    = (float*) alloc(GPAD * 256 * F);
    float*  q2      = (float*) alloc(GPAD * 256 * F);
    __bf16* q2b     = (__bf16*)alloc(GPAD * 256 * BF);
    float*  ffn1    = (float*) alloc(GPAD * 1024 * F);
    __bf16* ffn1b   = (__bf16*)alloc(GPAD * 1024 * BF);
    float*  ffn2    = (float*) alloc(GPAD * 256 * F);

    // 1) pack all weights to WMMA fragment layout
    launch_pack_w(vproj_w,  Wp_vproj, 256, 256,  stream);
    launch_pack_w(pw1,      Wp_pw1,   256, 256,  stream);
    launch_pack_w(pw2,      Wp_pw2,   256, 256,  stream);
    launch_pack_w(off_w,    Wp_off,   256, 256,  stream);
    launch_pack_w(attw_w,   Wp_attw,  256, 128,  stream);
    launch_pack_w(oproj_w,  Wp_oproj, 256, 256,  stream);
    launch_pack_w(sa_in_w,  Wp_sain,  256, 768,  stream);
    launch_pack_w(sa_out_w, Wp_saout, 256, 256,  stream);
    launch_pack_w(fw0,      Wp_fw0,   256, 1024, stream);
    launch_pack_w(fw1,      Wp_fw1,   1024, 256, stream);

    // 2) value path: A = flatten(value + value_pos) in bf16; val = A @ vproj + b
    {
        int total = MV * CDIM;
        build_av_kernel<<<(total + 255) / 256, 256, 0, stream>>>(
            feat[0], posf[0], feat[1], posf[1], feat[2], posf[2], feat[3], posf[3], Av);
    }
    launch_gemm(Av, Wp_vproj, vproj_b, val, MV, 256, 256, 0, stream);

    // 3) point pos-embed MLP
    {
        int total = 800 * 256;
        pos_l1_kernel<<<(total + 255) / 256, 256, 0, stream>>>(points, pw0, pb0, x1);
    }
    launch_pack_act(x1, x1b, 800, 800, 256, stream);
    launch_gemm(x1b, Wp_pw1, pb1, x2, 800, 256, 256, 1, stream);
    launch_pack_act(x2, x2b, 800, 800, 256, stream);
    launch_gemm(x2b, Wp_pw2, pb2, pos_emb, 800, 256, 256, 0, stream);

    // 4) q_in, offsets, attention weights
    {
        int total = GROWS * 256;
        build_qin_kernel<<<(total + 255) / 256, 256, 0, stream>>>(pos_emb, pc, gc, gp, q_in);
    }
    launch_pack_act(q_in, q_inb, GROWS, GPAD, 256, stream);
    launch_gemm(q_inb, Wp_off,  off_b,  offs,  GPAD, 256, 256, 0, stream);
    launch_gemm(q_inb, Wp_attw, attw_b, attwv, GPAD, 128, 256, 0, stream);
    {
        int total = GROWS * 8;
        softmax_attw_kernel<<<(total + 255) / 256, 256, 0, stream>>>(attwv);
    }

    // 5) deformable sampling + output projection + LN1
    {
        int total = GROWS * 256;
        sample_kernel<<<(total + 255) / 256, 256, 0, stream>>>(val, offs, attwv, points, msda);
    }
    launch_pack_act(msda, msdab, GROWS, GPAD, 256, stream);
    launch_gemm(msdab, Wp_oproj, oproj_b, oproj_o, GPAD, 256, 256, 0, stream);
    ln1_kernel<<<GROWS, 256, 0, stream>>>(oproj_o, q_in, pc, gc, ln1_g, ln1_b, q1);

    // 6) self-attention + LN2
    launch_pack_act(q1, q1b, GROWS, GPAD, 256, stream);
    launch_gemm(q1b, Wp_sain, sa_in_b, qkv, GPAD, 768, 256, 0, stream);
    self_attn_kernel<<<BATCH * 8, 128, 0, stream>>>(qkv, o_attn);
    launch_pack_act(o_attn, o_attnb, GROWS, GPAD, 256, stream);
    launch_gemm(o_attnb, Wp_saout, sa_out_b, sa_o, GPAD, 256, 256, 0, stream);
    ln_add_kernel<<<GROWS, 256, 0, stream>>>(q1, sa_o, ln2_g, ln2_b, q2);

    // 7) FFN + LN3 -> d_out (global token only)
    launch_pack_act(q2, q2b, GROWS, GPAD, 256, stream);
    launch_gemm(q2b, Wp_fw0, fb0, ffn1, GPAD, 1024, 256, 1, stream);
    launch_pack_act(ffn1, ffn1b, GROWS, GPAD, 1024, stream);
    launch_gemm(ffn1b, Wp_fw1, fb1, ffn2, GPAD, 256, 1024, 0, stream);
    ln3_out_kernel<<<BATCH, 256, 0, stream>>>(q2, ffn2, ln3_g, ln3_b, (float*)d_out);

    (void)in_sizes; (void)n_in; (void)out_size; (void)ws_size;
}
